// CausalSelfAttention_30545807409465
// MI455X (gfx1250) — matrix operand order repaired and checked by reading, not compiled, not run on hardware
//
#include <hip/hip_runtime.h>
#include <math.h>

// ---------------------------------------------------------------------------
// CDNA5 / gfx1250 causal self-attention, bf16 WMMA + async-LDS pipeline
//   B=4, T=2048, C=1024, H=16, D=64
//   prep   : x fp32->bf16 ; W_attn/W_proj fp32 -> bf16 transposed [N][K]
//   kernel1: QKV GEMM  (async double-buffered LDS, WMMA bf16) -> Q/K/V^T bf16
//   kernel2: flash attention: WMMA bf16, fused DPP max-reduce, row-sum WMMA
//   kernel3: proj GEMM (async double-buffered LDS, WMMA bf16) -> fp32 out
// ---------------------------------------------------------------------------

typedef __bf16 bf16;
typedef bf16  v16bf __attribute__((ext_vector_type(16)));
typedef float v8f   __attribute__((ext_vector_type(8)));

#define TDIM 2048
#define CDIM 1024
#define HEADS 16
#define HD 64
// 1/sqrt(64) * log2(e): softmax computed in base-2 domain
#define QSCALE 0.18033688011112042f

union V16U { uint4 u[2]; v16bf v; };

// A-matrix fragment (16x32 bf16): lane holds row M=lane%16.
// element e -> K = (e/8)*16 + g*8 + (e%8)   (g = lane/16)
__device__ __forceinline__ v16bf frag_a(const bf16* row, int g) {
  V16U t;
  t.u[0] = *reinterpret_cast<const uint4*>(row + g * 8);
  t.u[1] = *reinterpret_cast<const uint4*>(row + 16 + g * 8);
  return t.v;
}

// B-matrix fragment (32x16 bf16): lane holds col N=lane%16.
// element e -> K = (e/8)*16 + g*8 + (e%8)   (same as frag_a)
__device__ __forceinline__ v16bf frag_b(const bf16* colbase, int g) {
  V16U t;
  t.u[0] = *reinterpret_cast<const uint4*>(colbase + g * 8);
  t.u[1] = *reinterpret_cast<const uint4*>(colbase + 16 + g * 8);
  return t.v;
}

__device__ __forceinline__ v8f wmma_bf16(v16bf a, v16bf b, v8f c) {
  return __builtin_amdgcn_wmma_f32_16x16x32_bf16(
      false, a, false, b, (short)0, c, false, false);
}

// gfx1250 async global->LDS copy (ASYNCcnt-tracked).  l must be an LDS addr.
__device__ __forceinline__ void async_b128(const bf16* g, bf16* l) {
  asm volatile("global_load_async_to_lds_b128 %0, %1, off"
               :: "v"((unsigned)(uintptr_t)l), "v"(g) : "memory");
}
__device__ __forceinline__ void wait_async0() {
  asm volatile("s_wait_asynccnt 0x0" ::: "memory");
}

// ----- raw VALU max (no canonicalize) and fused DPP16 max-allreduce --------
__device__ __forceinline__ float fmax_raw(float a, float b) {
  float d;
  asm("v_max_num_f32 %0, %1, %2" : "=v"(d) : "v"(a), "v"(b));
  return d;
}
// max over the 16-lane row: 4 single-instruction DPP-fused max steps
__device__ __forceinline__ float rowmax16(float x) {
  float y;
  asm("v_max_num_f32_dpp %0, %1, %1 quad_perm:[1,0,3,2] row_mask:0xf bank_mask:0xf bound_ctrl:1"
      : "=v"(y) : "v"(x));
  asm("v_max_num_f32_dpp %0, %1, %1 quad_perm:[2,3,0,1] row_mask:0xf bank_mask:0xf bound_ctrl:1"
      : "=v"(x) : "v"(y));
  asm("v_max_num_f32_dpp %0, %1, %1 row_ror:4 row_mask:0xf bank_mask:0xf bound_ctrl:1"
      : "=v"(y) : "v"(x));
  asm("v_max_num_f32_dpp %0, %1, %1 row_ror:8 row_mask:0xf bank_mask:0xf bound_ctrl:1"
      : "=v"(x) : "v"(y));
  return x;
}

// ---------------------------------------------------------------------------
// prep kernels
// ---------------------------------------------------------------------------
__global__ void __launch_bounds__(256)
cvt_fp32_bf16(const float* __restrict__ s, bf16* __restrict__ d) {
  const size_t i = ((size_t)blockIdx.x * 256 + threadIdx.x) * 8;
  float4 f0 = reinterpret_cast<const float4*>(s + i)[0];
  float4 f1 = reinterpret_cast<const float4*>(s + i)[1];
  union { bf16 h[8]; uint4 u; } t;
  t.h[0] = (bf16)f0.x; t.h[1] = (bf16)f0.y; t.h[2] = (bf16)f0.z; t.h[3] = (bf16)f0.w;
  t.h[4] = (bf16)f1.x; t.h[5] = (bf16)f1.y; t.h[6] = (bf16)f1.z; t.h[7] = (bf16)f1.w;
  *reinterpret_cast<uint4*>(d + i) = t.u;
}

// W[K][N] fp32 -> Wt[N][K] bf16 (32x32 LDS tile transpose)
__global__ void __launch_bounds__(256)
transpose_cvt(const float* __restrict__ W, bf16* __restrict__ Wt, int K, int N) {
  __shared__ bf16 tile[32][33];
  const int tk = blockIdx.x * 32;
  const int tn = blockIdx.y * 32;
  const int c  = threadIdx.x & 31;
  const int rr = threadIdx.x >> 5;  // 0..7
#pragma unroll
  for (int i = 0; i < 32; i += 8)
    tile[rr + i][c] = (bf16)W[(size_t)(tk + rr + i) * N + tn + c];
  __syncthreads();
#pragma unroll
  for (int i = 0; i < 32; i += 8)
    Wt[(size_t)(tn + rr + i) * K + tk + c] = tile[c][rr + i];
}

// ---------------------------------------------------------------------------
// Double-buffered async WMMA GEMM: C[M,N] = A[M,K] * Bt[N,K]^T + bias
// A, Bt both bf16 row-major in K.  128x128 tile / 256 threads, K-step 64.
// MODE 0: scatter epilogue -> Q(*QSCALE)/K/V^T bf16.  MODE 1: fp32 out.
// ---------------------------------------------------------------------------
template <int MODE>
__global__ void __launch_bounds__(256)
gemm_bf16_async(const bf16* __restrict__ A, const bf16* __restrict__ Bt,
                const float* __restrict__ bias,
                bf16* __restrict__ outQ, bf16* __restrict__ outK,
                bf16* __restrict__ outVt, float* __restrict__ outC,
                int M, int N, int K)
{
  __shared__ alignas(16) bf16 Abuf[2][128 * 64];  // 16KB x2
  __shared__ alignas(16) bf16 Bbuf[2][128 * 64];  // 16KB x2

  const int tilesN = N >> 7;
  const int tileM = (blockIdx.x / tilesN) << 7;
  const int tileN = (blockIdx.x % tilesN) << 7;

  const int tid  = threadIdx.x;
  const int w    = tid >> 5;
  const int lane = tid & 31;
  const int g    = lane >> 4;
  const int r    = lane & 15;
  const int wm   = (w >> 2) * 64;  // 0 / 64
  const int wn   = (w & 3) * 32;   // 0/32/64/96

  const int srow = tid >> 1;        // 0..127
  const int scol = (tid & 1) * 32;  // bf16 elems

  const bf16* gA = A  + (size_t)(tileM + srow) * K + scol;
  const bf16* gB = Bt + (size_t)(tileN + srow) * K + scol;

  v8f acc[4][2];
#pragma unroll
  for (int i = 0; i < 4; ++i)
#pragma unroll
    for (int j = 0; j < 2; ++j)
      acc[i][j] = v8f{0.f, 0.f, 0.f, 0.f, 0.f, 0.f, 0.f, 0.f};

#pragma unroll
  for (int i = 0; i < 4; ++i) {
    async_b128(gA + i * 8, &Abuf[0][srow * 64 + scol + i * 8]);
    async_b128(gB + i * 8, &Bbuf[0][srow * 64 + scol + i * 8]);
  }

  const int steps = K >> 6;
  for (int s = 0; s < steps; ++s) {
    const int cur = s & 1;
    wait_async0();     // my async copies into buf[cur] have landed
    __syncthreads();   // everyone's landed; prior reads of buf[cur^1] done

    if (s + 1 < steps) {
      const int k0n = (s + 1) << 6;
#pragma unroll
      for (int i = 0; i < 4; ++i) {
        async_b128(gA + k0n + i * 8, &Abuf[cur ^ 1][srow * 64 + scol + i * 8]);
        async_b128(gB + k0n + i * 8, &Bbuf[cur ^ 1][srow * 64 + scol + i * 8]);
      }
    }

    const bf16* Ab = &Abuf[cur][0];
    const bf16* Bb = &Bbuf[cur][0];
#pragma unroll
    for (int kk = 0; kk < 2; ++kk) {
      v16bf af[4];
#pragma unroll
      for (int mr = 0; mr < 4; ++mr)
        af[mr] = frag_a(Ab + (wm + mr * 16 + r) * 64 + kk * 32, g);
      v16bf bfr[2];
#pragma unroll
      for (int nc = 0; nc < 2; ++nc)
        bfr[nc] = frag_b(Bb + (wn + nc * 16 + r) * 64 + kk * 32, g);
#pragma unroll
      for (int mr = 0; mr < 4; ++mr)
#pragma unroll
        for (int nc = 0; nc < 2; ++nc)
          acc[mr][nc] = wmma_bf16(af[mr], bfr[nc], acc[mr][nc]);
    }
  }

  // ---- epilogue (C/D layout: row = base + g*8 + v, col = tile + r) ----
#pragma unroll
  for (int nc = 0; nc < 2; ++nc) {
    const int col = tileN + wn + nc * 16 + r;
    const float bv = bias[col];
#pragma unroll
    for (int mr = 0; mr < 4; ++mr) {
      const int rbase = tileM + wm + mr * 16 + g * 8;
#pragma unroll
      for (int v = 0; v < 8; ++v) {
        const int row = rbase + v;
        const float val = acc[mr][nc][v] + bv;
        if (MODE == 0) {
          const int which = col >> 10;          // 0=q 1=k 2=v
          const int c = col & (CDIM - 1);
          const int h = c >> 6;
          const int d = c & (HD - 1);
          const int b = row >> 11;
          const int t = row & (TDIM - 1);
          const size_t bh = (size_t)b * HEADS + h;
          if (which == 0)
            outQ[(bh * TDIM + t) * HD + d] = (bf16)(val * QSCALE);
          else if (which == 1)
            outK[(bh * TDIM + t) * HD + d] = (bf16)val;
          else
            outVt[(bh * HD + d) * TDIM + t] = (bf16)val;            // transposed
        } else {
          outC[(size_t)row * N + col] = val;
        }
      }
    }
  }
}

// ---------------------------------------------------------------------------
// Flash attention: grid (B*H, T/128), 8 wave32 per block, 16 q rows / wave.
// Unmasked main loop + masked diagonal tail.  Row max via fused DPP max;
// row sum via a 5th WMMA accumulator against a constant B-fragment.
// ---------------------------------------------------------------------------
__global__ void __launch_bounds__(256)
flash_attn_wmma(const bf16* __restrict__ Qb, const bf16* __restrict__ Kb,
                const bf16* __restrict__ Vt, bf16* __restrict__ Y)
{
  __shared__ alignas(16) bf16 Pl[8][16 * 32];

  const int bh   = blockIdx.x;
  const int qb   = blockIdx.y;
  const int w    = threadIdx.x >> 5;
  const int lane = threadIdx.x & 31;
  const int g    = lane >> 4;
  const int r    = lane & 15;
  const int r0   = qb * 128 + w * 16;

  const bf16* qrow = Qb + ((size_t)bh * TDIM + r0 + r) * HD;
  const v16bf qf0 = frag_a(qrow, g);
  const v16bf qf1 = frag_a(qrow + 32, g);

  // constant B-fragment: column 0 all ones, other columns zero
  V16U sf;
  {
    const unsigned fill = (r == 0) ? 0x3F803F80u : 0u;  // packed bf16 1.0
    sf.u[0] = uint4{fill, fill, fill, fill};
    sf.u[1] = uint4{fill, fill, fill, fill};
  }
  const v16bf sumfrag = sf.v;

  const v8f zacc = v8f{0.f, 0.f, 0.f, 0.f, 0.f, 0.f, 0.f, 0.f};
  v8f o[5];                       // o[0..3] = output d-tiles, o[4] = row sums
#pragma unroll
  for (int dt = 0; dt < 5; ++dt) o[dt] = zacc;

  float mrow[8];
#pragma unroll
  for (int v = 0; v < 8; ++v) mrow[v] = -__builtin_inff();

  auto tile = [&](int kb, bool masked) __attribute__((always_inline)) {
    v8f s[2];
#pragma unroll
    for (int nt = 0; nt < 2; ++nt) {
      const bf16* krow = Kb + ((size_t)bh * TDIM + kb + nt * 16 + r) * HD;
      const v16bf kf0 = frag_b(krow, g);
      const v16bf kf1 = frag_b(krow + 32, g);
      s[nt] = wmma_bf16(qf0, kf0, zacc);
      s[nt] = wmma_bf16(qf1, kf1, s[nt]);
    }

#pragma unroll
    for (int v = 0; v < 8; ++v) {
      const int row = r0 + g * 8 + v;
      if (masked) {
#pragma unroll
        for (int nt = 0; nt < 2; ++nt)
          if (kb + nt * 16 + r > row) s[nt][v] = -__builtin_inff();
      }
      const float mx = rowmax16(fmax_raw(s[0][v], s[1][v]));
      const float mnew  = fmax_raw(mrow[v], mx);
      const float alpha = __builtin_amdgcn_exp2f(mrow[v] - mnew);
      const float p0 = __builtin_amdgcn_exp2f(s[0][v] - mnew);
      const float p1 = __builtin_amdgcn_exp2f(s[1][v] - mnew);
      mrow[v] = mnew;
#pragma unroll
      for (int dt = 0; dt < 5; ++dt) o[dt][v] *= alpha;

      Pl[w][(g * 8 + v) * 32 + r]      = (bf16)p0;
      Pl[w][(g * 8 + v) * 32 + 16 + r] = (bf16)p1;
    }

    asm volatile("s_wait_dscnt 0x0" ::: "memory");  // wave-local LDS round-trip

    const v16bf pf = frag_a(&Pl[w][r * 32], g);
#pragma unroll
    for (int dt = 0; dt < 4; ++dt) {
      const bf16* vrow = Vt + ((size_t)bh * HD + dt * 16 + r) * TDIM + kb;
      const v16bf vf = frag_b(vrow, g);
      o[dt] = wmma_bf16(pf, vf, o[dt]);
    }
    o[4] = wmma_bf16(pf, sumfrag, o[4]);   // running row sums
  };

  // full (mask-free) tiles: kb+31 <= r0
  const int full_end = (r0 >= 31) ? ((((r0 - 31) >> 5) + 1) << 5) : 0;
  int kb = 0;
  for (; kb < full_end; kb += 32) tile(kb, false);
  for (; kb < r0 + 16; kb += 32) tile(kb, true);

  const int b = bh >> 4;
  const int h = bh & (HEADS - 1);
#pragma unroll
  for (int v = 0; v < 8; ++v) {
    const int row = r0 + g * 8 + v;
    const float lsum = __shfl(o[4][v], g << 4);   // row sum lives in col 0
    const float inv = 1.0f / lsum;
    bf16* yrow = Y + ((size_t)b * TDIM + row) * CDIM + h * HD + r;
#pragma unroll
    for (int dt = 0; dt < 4; ++dt)
      yrow[dt * 16] = (bf16)(o[dt][v] * inv);
  }
}

// ---------------------------------------------------------------------------
extern "C" void kernel_launch(void* const* d_in, const int* in_sizes, int n_in,
                              void* d_out, int out_size, void* d_ws, size_t ws_size,
                              hipStream_t stream)
{
  (void)in_sizes; (void)n_in; (void)out_size; (void)ws_size;

  const float* x      = (const float*)d_in[0];
  const float* W_attn = (const float*)d_in[1];
  const float* b_attn = (const float*)d_in[2];
  const float* W_proj = (const float*)d_in[3];
  const float* b_proj = (const float*)d_in[4];
  float* out = (float*)d_out;

  const size_t ELEMS = (size_t)4 * TDIM * CDIM;   // 8,388,608
  const size_t MB = 1024 * 1024;
  char* ws = (char*)d_ws;
  bf16* Xb  = (bf16*)(ws);             // [8192,1024]
  bf16* Qb  = (bf16*)(ws + 16 * MB);   // [B,H,T,D] pre-scaled by QSCALE
  bf16* Kb  = (bf16*)(ws + 32 * MB);   // [B,H,T,D]
  bf16* Vt  = (bf16*)(ws + 48 * MB);   // [B,H,D,T]
  bf16* Y   = (bf16*)(ws + 64 * MB);   // [B,T,C]
  bf16* WtA = (bf16*)(ws + 80 * MB);   // [3072,1024]
  bf16* WtP = (bf16*)(ws + 86 * MB);   // [1024,1024]

  const int M = 4 * TDIM;  // 8192

  cvt_fp32_bf16<<<dim3(ELEMS / (256 * 8)), 256, 0, stream>>>(x, Xb);
  transpose_cvt<<<dim3(CDIM / 32, 3 * CDIM / 32), 256, 0, stream>>>(W_attn, WtA, CDIM, 3 * CDIM);
  transpose_cvt<<<dim3(CDIM / 32, CDIM / 32), 256, 0, stream>>>(W_proj, WtP, CDIM, CDIM);

  gemm_bf16_async<0><<<dim3((M / 128) * (3 * CDIM / 128)), 256, 0, stream>>>(
      Xb, WtA, b_attn, Qb, Kb, Vt, nullptr, M, 3 * CDIM, CDIM);

  flash_attn_wmma<<<dim3(4 * HEADS, TDIM / 128), 256, 0, stream>>>(Qb, Kb, Vt, Y);

  gemm_bf16_async<1><<<dim3((M / 128) * (CDIM / 128)), 256, 0, stream>>>(
      Y, WtP, b_proj, nullptr, nullptr, nullptr, out, M, CDIM, CDIM);
}
